// CausalSelfAttention_32882269618637
// MI455X (gfx1250) — matrix operand order repaired and checked by reading, not compiled, not run on hardware
//
#include <hip/hip_runtime.h>
#include <hip/hip_bf16.h>

typedef __attribute__((ext_vector_type(16))) _Float16 v16h;
typedef __attribute__((ext_vector_type(8)))  float    v8f;
typedef __attribute__((ext_vector_type(4)))  int      v4i;

#define TT 2048
#define CC 1024
#define HH 16
#define DD 64
#define BB 2
#define MM (BB * TT)      // 4096 rows of x
#define N3 (3 * CC)       // 3072

union Frag {
    v16h h;
    uint4 u[2];
};

static __device__ inline v8f wmma16(v16h a, v16h b, v8f c) {
    // D = A(16x32 f16) * B(32x16 f16) + C(16x16 f32)
    return __builtin_amdgcn_wmma_f32_16x16x32_f16(false, a, false, b, (short)0, c, false, false);
}

#if __has_builtin(__builtin_amdgcn_global_load_async_to_lds_b128)
#define HAS_ASYNC_LDS 1
static __device__ inline void async_b128(const _Float16* g, _Float16* l) {
    __builtin_amdgcn_global_load_async_to_lds_b128(
        (__attribute__((address_space(1))) v4i*)g,
        (__attribute__((address_space(3))) v4i*)l,
        0, 0);
}
#else
#define HAS_ASYNC_LDS 0
#endif

__global__ void cvt_f32_f16(const float* __restrict__ in, _Float16* __restrict__ out, int n) {
    int i = blockIdx.x * blockDim.x + threadIdx.x;
    if (i < n) out[i] = (_Float16)in[i];
}

// Cooperative tile movement helpers. Per thread: A tile 64x64 halves -> 2
// b128, B tile 128x64 halves -> 4 b128. Named scalar staging (no arrays) so
// the fallback path keeps everything in VGPRs.
#define GLOAD(k0)                                                             \
    do {                                                                      \
        const _Float16* ap_ = Ag + (size_t)lr * K + (k0) + lc;                \
        const _Float16* bp_ = Bg + (size_t)lr * K + (k0) + lc;                \
        a0 = *(const uint4*)ap_;            a1 = *(const uint4*)(ap_ + 32 * K); \
        b0 = *(const uint4*)bp_;            b1 = *(const uint4*)(bp_ + 32 * K); \
        b2 = *(const uint4*)(bp_ + 64 * K); b3 = *(const uint4*)(bp_ + 96 * K); \
    } while (0)

#define SSTORE(bf)                                                            \
    do {                                                                      \
        _Float16* as_ = &As[bf][lr * 72 + lc];                                \
        _Float16* bs_ = &Bs[bf][lr * 72 + lc];                                \
        *(uint4*)as_ = a0;             *(uint4*)(as_ + 32 * 72) = a1;         \
        *(uint4*)bs_ = b0;             *(uint4*)(bs_ + 32 * 72) = b1;         \
        *(uint4*)(bs_ + 64 * 72) = b2; *(uint4*)(bs_ + 96 * 72) = b3;         \
    } while (0)

#if HAS_ASYNC_LDS
#define AISSUE(k0, bf)                                                        \
    do {                                                                      \
        const _Float16* ap_ = Ag + (size_t)lr * K + (k0) + lc;                \
        const _Float16* bp_ = Bg + (size_t)lr * K + (k0) + lc;                \
        _Float16* as_ = &As[bf][lr * 72 + lc];                                \
        _Float16* bs_ = &Bs[bf][lr * 72 + lc];                                \
        async_b128(ap_,          as_);                                        \
        async_b128(ap_ + 32 * K, as_ + 32 * 72);                              \
        async_b128(bp_,          bs_);                                        \
        async_b128(bp_ + 32 * K, bs_ + 32 * 72);                              \
        async_b128(bp_ + 64 * K, bs_ + 64 * 72);                              \
        async_b128(bp_ + 96 * K, bs_ + 96 * 72);                              \
    } while (0)
#define AWAIT() asm volatile("s_wait_asynccnt 0x0" ::: "memory")
#endif

// 8-WMMA compute step on LDS buffer `bf` (two 32-K substeps, 4 N-tiles).
#define COMPUTE(bf)                                                           \
    do {                                                                      \
        const _Float16* as_ = As[bf];                                         \
        const _Float16* bs_ = Bs[bf];                                         \
        for (int ks = 0; ks < 64; ks += 32) {                                 \
            Frag a_;                                                          \
            int arow_ = wm * 16 + r;                                          \
            a_.u[0] = *(const uint4*)(as_ + arow_ * 72 + ks + hi * 8);        \
            a_.u[1] = *(const uint4*)(as_ + arow_ * 72 + ks + hi * 8 + 16);   \
            for (int nt = 0; nt < 4; ++nt) {                                  \
                Frag b_;                                                      \
                int brow_ = wn * 64 + nt * 16 + r;                            \
                b_.u[0] = *(const uint4*)(bs_ + brow_ * 72 + ks + hi * 8);    \
                b_.u[1] = *(const uint4*)(bs_ + brow_ * 72 + ks + hi * 8 + 16); \
                acc[nt] = wmma16(a_.h, b_.h, acc[nt]);                        \
            }                                                                 \
        }                                                                     \
    } while (0)

// ---------------------------------------------------------------------------
// QKV GEMM: out[m,n] = x[m,:] . W_attn[n,:] + b_attn[n]
// Block tile 64(M) x 128(N), K-step 64, double-buffered LDS fed by async
// global->LDS copies (fallback: VGPR staging). 8 waves, 16x64 register tile
// per wave. Scatter: q (pre-scaled 1/8) / k as [B,H,T,D], v transposed
// [B,H,D,T] f16.
// ---------------------------------------------------------------------------
__global__ void __launch_bounds__(256) qkv_gemm(const _Float16* __restrict__ A,
                                                const _Float16* __restrict__ W,
                                                const float*    __restrict__ bias,
                                                _Float16* __restrict__ qh,
                                                _Float16* __restrict__ kh,
                                                _Float16* __restrict__ vt) {
    __shared__ __align__(16) _Float16 As[2][64 * 72];
    __shared__ __align__(16) _Float16 Bs[2][128 * 72];
    const int K  = CC;
    const int NB = N3 / 128;                      // 24 N-tiles
    int tid  = threadIdx.x;
    int lane = tid & 31;
    int wave = tid >> 5;
    int wm = wave & 3;
    int wn = wave >> 2;
    int m0 = (blockIdx.x / NB) * 64;
    int n0 = (blockIdx.x % NB) * 128;
    int r  = lane & 15;
    int hi = lane >> 4;
    int lr = tid >> 3;            // 0..31
    int lc = (tid & 7) * 8;       // 0..56
    const _Float16* Ag = A + (size_t)m0 * K;
    const _Float16* Bg = W + (size_t)n0 * K;

    v8f acc[4] = {};
    int buf = 0;
#if HAS_ASYNC_LDS
    AISSUE(0, 0);
    AWAIT();
    __syncthreads();
    for (int k0 = 0; k0 < K; k0 += 64) {
        bool hasnext = (k0 + 64 < K);
        if (hasnext) AISSUE(k0 + 64, buf ^ 1);
        COMPUTE(buf);
        if (hasnext) {
            AWAIT();
            __syncthreads();
            buf ^= 1;
        }
    }
#else
    uint4 a0, a1, b0, b1, b2, b3;
    GLOAD(0);
    SSTORE(0);
    __syncthreads();
    for (int k0 = 0; k0 < K; k0 += 64) {
        bool hasnext = (k0 + 64 < K);
        if (hasnext) {
            GLOAD(k0 + 64);
            if (k0 + 128 < K) {
                __builtin_prefetch(Ag + (size_t)lr * K + k0 + 128 + lc, 0, 1);
                __builtin_prefetch(Bg + (size_t)lr * K + k0 + 128 + lc, 0, 1);
            }
        }
        COMPUTE(buf);
        if (hasnext) {
            SSTORE(buf ^ 1);
            __syncthreads();
            buf ^= 1;
        }
    }
#endif

    for (int nt = 0; nt < 4; ++nt) {
        int n = n0 + wn * 64 + nt * 16 + r;
        float bv = bias[n];
        int s = n / CC;
        int c = n % CC;
        int h = c / DD;
        int d = c % DD;
        for (int v = 0; v < 8; ++v) {             // D-layout: row = v + hi*8
            int m = m0 + wm * 16 + hi * 8 + v;
            int bidx = m / TT, t = m % TT;
            float val = acc[nt][v] + bv;
            if (s == 0)
                qh[((size_t)(bidx * HH + h) * TT + t) * DD + d] = (_Float16)(val * 0.125f);
            else if (s == 1)
                kh[((size_t)(bidx * HH + h) * TT + t) * DD + d] = (_Float16)val;
            else
                vt[((size_t)(bidx * HH + h) * DD + d) * TT + t] = (_Float16)val;
        }
    }
}

// ---------------------------------------------------------------------------
// Flash attention: one wave per 16 query rows of one (b,h).
// Key blocks of 32: S = Q(16x64) @ K^T (4 WMMA), online softmax in C-layout,
// P re-laid out via per-wave LDS, O += P(16x32) @ V(32x64) (4 WMMA).
// V stored transposed [B,H,D,T]: every V fragment = two contiguous b128 loads.
// ---------------------------------------------------------------------------
__global__ void __launch_bounds__(256) attn_kernel(const _Float16* __restrict__ qh,
                                                   const _Float16* __restrict__ kh,
                                                   const _Float16* __restrict__ vt,
                                                   _Float16* __restrict__ yh) {
    __shared__ __align__(16) _Float16 Ps[8][16 * 40];   // per-wave P tile
    int lane = threadIdx.x & 31;
    int wv   = threadIdx.x >> 5;
    int w    = blockIdx.x * 8 + wv;
    int bh   = w >> 7;                 // / (T/16)
    int q0   = (w & 127) << 4;
    int r    = lane & 15;
    int hi   = lane >> 4;
    const _Float16* Q  = qh + (size_t)bh * TT * DD;
    const _Float16* Km = kh + (size_t)bh * TT * DD;
    const _Float16* Vt = vt + (size_t)bh * DD * TT;
    _Float16* ps = Ps[wv];

    Frag qa[2];                        // Q rows q0..q0+15, A-layout, d=0..63
    {
        const _Float16* qrow = Q + (size_t)(q0 + r) * DD;
        for (int kk = 0; kk < 2; ++kk) {
            qa[kk].u[0] = *(const uint4*)(qrow + kk * 32 + hi * 8);
            qa[kk].u[1] = *(const uint4*)(qrow + kk * 32 + hi * 8 + 16);
        }
    }

    v8f o[4] = {};
    float mrow[8], lrow[8];
    for (int v = 0; v < 8; ++v) { mrow[v] = -1e30f; lrow[v] = 0.0f; }

    int jend = q0 + 16;
    for (int j0 = 0; j0 < jend; j0 += 32) {
        v8f s[2] = {};
        for (int jj = 0; jj < 2; ++jj) {           // keys j0+jj*16+r (B-layout)
            const _Float16* krow = Km + (size_t)(j0 + jj * 16 + r) * DD;
            for (int kk = 0; kk < 2; ++kk) {
                Frag kb;
                kb.u[0] = *(const uint4*)(krow + kk * 32 + hi * 8);
                kb.u[1] = *(const uint4*)(krow + kk * 32 + hi * 8 + 16);
                s[jj] = wmma16(qa[kk].h, kb.h, s[jj]);
            }
        }
        // causal mask + online softmax; row of component v is q0 + hi*8 + v
        for (int v = 0; v < 8; ++v) {
            int qrow = q0 + hi * 8 + v;
            float s0 = s[0][v], s1 = s[1][v];
            if (j0 + r      > qrow) s0 = -1e30f;
            if (j0 + 16 + r > qrow) s1 = -1e30f;
            float mx = fmaxf(s0, s1);
            mx = fmaxf(mx, __shfl_xor(mx, 1, 32));
            mx = fmaxf(mx, __shfl_xor(mx, 2, 32));
            mx = fmaxf(mx, __shfl_xor(mx, 4, 32));
            mx = fmaxf(mx, __shfl_xor(mx, 8, 32));
            float mnew = fmaxf(mrow[v], mx);
            float p0 = __expf(s0 - mnew);
            float p1 = __expf(s1 - mnew);
            float rs = p0 + p1;
            rs += __shfl_xor(rs, 1, 32);
            rs += __shfl_xor(rs, 2, 32);
            rs += __shfl_xor(rs, 4, 32);
            rs += __shfl_xor(rs, 8, 32);
            float alpha = __expf(mrow[v] - mnew);
            mrow[v] = mnew;
            lrow[v] = lrow[v] * alpha + rs;
            for (int dd = 0; dd < 4; ++dd) o[dd][v] *= alpha;
            int prow = hi * 8 + v;                 // stage P (row-major in LDS)
            ps[prow * 40 + r]      = (_Float16)p0;
            ps[prow * 40 + 16 + r] = (_Float16)p1;
        }
        Frag pa;                                   // reload P in A-layout
        pa.u[0] = *(const uint4*)(ps + r * 40 + hi * 8);
        pa.u[1] = *(const uint4*)(ps + r * 40 + hi * 8 + 16);
        for (int dd = 0; dd < 4; ++dd) {           // O += P @ V
            const _Float16* vrow = Vt + (size_t)(dd * 16 + r) * TT + j0;
            Frag vb;
            vb.u[0] = *(const uint4*)(vrow + hi * 8);
            vb.u[1] = *(const uint4*)(vrow + hi * 8 + 16);
            o[dd] = wmma16(pa.h, vb.h, o[dd]);
        }
    }

    int b = bh >> 4, hhead = bh & 15;
    for (int v = 0; v < 8; ++v) {
        int t = q0 + hi * 8 + v;
        float inv = 1.0f / lrow[v];
        for (int dd = 0; dd < 4; ++dd) {
            int d = dd * 16 + r;
            yh[(size_t)(b * TT + t) * CC + hhead * DD + d] = (_Float16)(o[dd][v] * inv);
        }
    }
}

// ---------------------------------------------------------------------------
// Output projection: out[m,n] = y[m,:] . W_proj[n,:] + b_proj[n]  (f32 out)
// Same 64x128 double-buffered async tiling as qkv_gemm.
// ---------------------------------------------------------------------------
__global__ void __launch_bounds__(256) proj_gemm(const _Float16* __restrict__ A,
                                                 const _Float16* __restrict__ W,
                                                 const float*    __restrict__ bias,
                                                 float* __restrict__ out) {
    __shared__ __align__(16) _Float16 As[2][64 * 72];
    __shared__ __align__(16) _Float16 Bs[2][128 * 72];
    const int K  = CC;
    const int NB = CC / 128;                      // 8 N-tiles
    int tid  = threadIdx.x;
    int lane = tid & 31;
    int wave = tid >> 5;
    int wm = wave & 3;
    int wn = wave >> 2;
    int m0 = (blockIdx.x / NB) * 64;
    int n0 = (blockIdx.x % NB) * 128;
    int r  = lane & 15;
    int hi = lane >> 4;
    int lr = tid >> 3;
    int lc = (tid & 7) * 8;
    const _Float16* Ag = A + (size_t)m0 * K;
    const _Float16* Bg = W + (size_t)n0 * K;

    v8f acc[4] = {};
    int buf = 0;
#if HAS_ASYNC_LDS
    AISSUE(0, 0);
    AWAIT();
    __syncthreads();
    for (int k0 = 0; k0 < K; k0 += 64) {
        bool hasnext = (k0 + 64 < K);
        if (hasnext) AISSUE(k0 + 64, buf ^ 1);
        COMPUTE(buf);
        if (hasnext) {
            AWAIT();
            __syncthreads();
            buf ^= 1;
        }
    }
#else
    uint4 a0, a1, b0, b1, b2, b3;
    GLOAD(0);
    SSTORE(0);
    __syncthreads();
    for (int k0 = 0; k0 < K; k0 += 64) {
        bool hasnext = (k0 + 64 < K);
        if (hasnext) {
            GLOAD(k0 + 64);
            if (k0 + 128 < K) {
                __builtin_prefetch(Ag + (size_t)lr * K + k0 + 128 + lc, 0, 1);
                __builtin_prefetch(Bg + (size_t)lr * K + k0 + 128 + lc, 0, 1);
            }
        }
        COMPUTE(buf);
        if (hasnext) {
            SSTORE(buf ^ 1);
            __syncthreads();
            buf ^= 1;
        }
    }
#endif

    for (int nt = 0; nt < 4; ++nt) {
        int n = n0 + wn * 64 + nt * 16 + r;
        float bv = bias[n];
        for (int v = 0; v < 8; ++v) {
            int m = m0 + wm * 16 + hi * 8 + v;
            out[(size_t)m * CC + n] = acc[nt][v] + bv;
        }
    }
}

extern "C" void kernel_launch(void* const* d_in, const int* in_sizes, int n_in,
                              void* d_out, int out_size, void* d_ws, size_t ws_size,
                              hipStream_t stream) {
    (void)in_sizes; (void)n_in; (void)out_size; (void)ws_size;
    const float* x      = (const float*)d_in[0];
    const float* W_attn = (const float*)d_in[1];
    const float* b_attn = (const float*)d_in[2];
    const float* W_proj = (const float*)d_in[3];
    const float* b_proj = (const float*)d_in[4];
    float* out = (float*)d_out;

    char* ws = (char*)d_ws;
    _Float16* xh  = (_Float16*)ws; ws += (size_t)MM * CC * 2;   // 8 MB
    _Float16* Wah = (_Float16*)ws; ws += (size_t)N3 * CC * 2;   // 6 MB
    _Float16* Wph = (_Float16*)ws; ws += (size_t)CC * CC * 2;   // 2 MB
    _Float16* qh  = (_Float16*)ws; ws += (size_t)MM * CC * 2;   // 8 MB
    _Float16* kh  = (_Float16*)ws; ws += (size_t)MM * CC * 2;   // 8 MB
    _Float16* vt  = (_Float16*)ws; ws += (size_t)MM * CC * 2;   // 8 MB (transposed)
    _Float16* yh  = (_Float16*)ws;                              // 8 MB

    int nx = MM * CC, nwa = N3 * CC, nwp = CC * CC;
    cvt_f32_f16<<<(nx  + 255) / 256, 256, 0, stream>>>(x,      xh,  nx);
    cvt_f32_f16<<<(nwa + 255) / 256, 256, 0, stream>>>(W_attn, Wah, nwa);
    cvt_f32_f16<<<(nwp + 255) / 256, 256, 0, stream>>>(W_proj, Wph, nwp);

    qkv_gemm<<<(MM / 64) * (N3 / 128), 256, 0, stream>>>(xh, Wah, b_attn, qh, kh, vt);
    attn_kernel<<<(BB * HH * (TT / 16)) / 8, 256, 0, stream>>>(qh, kh, vt, yh);
    proj_gemm<<<(MM / 64) * (CC / 128), 256, 0, stream>>>(yh, Wph, b_proj, out);
}
